// SIM3Policy_73796128080448
// MI455X (gfx1250) — compile-verified, matrix operand order checked by reading, not run on hardware
//
#include <hip/hip_runtime.h>
#include <hip/hip_bf16.h>
#include <math.h>

typedef __attribute__((ext_vector_type(16))) _Float16 v16h;
typedef __attribute__((ext_vector_type(8)))  _Float16 v8h;
typedef __attribute__((ext_vector_type(8)))  float    v8f;

#define EPSF 1e-6f
#define Bn   8
#define Mn   16384
#define Cn   128
#define Hn   128
#define VIN  136
#define K1PAD 160      // VIN padded to multiple of 32
#define PTS  32        // points per workgroup tile
#define NCOL 96        // 3 * PTS columns
#define NT   6         // NCOL / 16
#define KSTR 168       // LDS activation row stride (halves), 16B aligned, bank-spread
#define QSTR 130       // LDS qbuf row stride (floats)

// ---- d_ws layout (f16 element offsets) ----
#define W1H_OFF   0            // [128][160]
#define W1DH_OFF  20480        // [128][128]
#define W2H_OFF   36864        // [128][128]
#define W2DH_OFF  53248        // [128][128]
#define WS_HALVES 69632

// ---- dynamic LDS layout for main kernel (bytes) ----
#define ACT_OFF   0
#define ACT_BYTES (NCOL * KSTR * 2)          // 32256
#define QBUF_OFF  ACT_BYTES
#define QBUF_BYTES (NCOL * QSTR * 4)         // 49920
#define S2V_OFF   (QBUF_OFF + QBUF_BYTES)    // 82176
#define W3_OFF    (S2V_OFF + 128 * 4)        // 82688
#define LDS_TOTAL (W3_OFF + 256 * 4)         // 83712  (< 320KB WGP LDS)

// ---- output layout (floats) ----
#define OFF_GRIP (Bn * (size_t)Mn * 6)       // 786432
#define OFF_AC   (OFF_GRIP + Bn * 2)         // 786448

// =====================================================================
// Kernel 0: convert weights f32 -> f16 into d_ws (K-padded where needed)
// =====================================================================
__global__ __launch_bounds__(256) void sim3_prep_weights(
    const float* __restrict__ W1x,   // (128,136)
    const float* __restrict__ W1x_d, // (128,128)
    const float* __restrict__ W2x,   // (128,128)
    const float* __restrict__ W2x_d, // (128,128)
    _Float16* __restrict__ ws) {
  int idx = blockIdx.x * blockDim.x + threadIdx.x;
  if (idx >= WS_HALVES) return;
  float v;
  if (idx < W1DH_OFF) {                 // W1x padded to [128][160]
    int o = idx / K1PAD, k = idx % K1PAD;
    v = (k < VIN) ? W1x[o * VIN + k] : 0.0f;
  } else if (idx < W2H_OFF) {
    v = W1x_d[idx - W1DH_OFF];
  } else if (idx < W2DH_OFF) {
    v = W2x[idx - W2H_OFF];
  } else {
    v = W2x_d[idx - W2DH_OFF];
  }
  ws[idx] = (_Float16)v;
}

// =====================================================================
// WMMA GEMM over one workgroup tile:
//   out[o][n] = sum_k W[o][k] * act[n][k],  o in [0,128), n in [0,NCOL)
// wave id == M tile (8 waves == 8 M tiles). Each wave loads its A
// fragments (global f16 weights) ONCE into registers, then sweeps the
// 6 N tiles with LDS-only B-fragment loads + v_wmma.
// 16-bit A fragment lane layout (ISA 7.12.2): lanes 0-15 row M=lane,
// halves K = 0..7 then 16..23 ; lanes 16-31 row M=lane-16, K = 8..15, 24..31.
// B assumed mirrored with N in place of M.
// =====================================================================
template <int KT>
__device__ __forceinline__ void gemm_tile(
    const _Float16* __restrict__ W, const int Kpad,
    const _Float16* act, float* qbuf, int tid) {
  const int mt    = tid >> 5;        // wave id == output-row tile
  const int lane  = tid & 31;
  const int lhalf = lane & 15;
  const int hi    = (lane >> 4) & 1;
  const int kOffA = hi ? 8 : 0;
  const int kOffB = hi ? 24 : 16;

  const _Float16* wrow = W + (mt * 16 + lhalf) * Kpad;
  v16h a[KT];
#pragma unroll
  for (int kt = 0; kt < KT; ++kt) {
    *(v8h*)&a[kt]        = *(const v8h*)(wrow + kt * 32 + kOffA);
    *((v8h*)&a[kt] + 1)  = *(const v8h*)(wrow + kt * 32 + kOffB);
  }

#pragma unroll
  for (int nt = 0; nt < NT; ++nt) {
    const _Float16* brow = act + (nt * 16 + lhalf) * KSTR;
    v8f acc = {};
#pragma unroll
    for (int kt = 0; kt < KT; ++kt) {
      v16h bf;
      *(v8h*)&bf       = *(const v8h*)(brow + kt * 32 + kOffA);
      *((v8h*)&bf + 1) = *(const v8h*)(brow + kt * 32 + kOffB);
      acc = __builtin_amdgcn_wmma_f32_16x16x32_f16(
          false, a[kt], false, bf, (short)0, acc, false, false);
    }
    // C/D layout: VGPR r -> row M = mt*16 + r + (hi?8:0), col N = nt*16+lhalf
    const int n = nt * 16 + lhalf;
    float* qn = qbuf + n * QSTR + mt * 16 + (hi ? 8 : 0);
#pragma unroll
    for (int r = 0; r < 8; ++r) qn[r] = acc[r];
  }
}

// =====================================================================
// Kernel 1: fused 4-layer vector-neuron MLP over point tiles
// =====================================================================
extern __shared__ char smem_raw[];

__global__ __launch_bounds__(256) void sim3_points_kernel(
    const float* __restrict__ point_feat,  // (B,C,3,M)
    const float* __restrict__ z_pos,       // (B,2,3)
    const float* __restrict__ z_dir,       // (B,6,3)
    const float* __restrict__ z_scalar,    // (B,2)
    const float* __restrict__ center,      // (B,1,3)
    const float* __restrict__ scale,       // (B,)
    const float* __restrict__ W1x_s,       // (128,2)
    const float* __restrict__ W3x,         // (2,128)
    const _Float16* __restrict__ ws,       // prepped f16 weights
    float* __restrict__ out) {             // offsets at [0, B*M*6)
  const int tid  = threadIdx.x;
  const int b    = blockIdx.y;
  const int m0   = blockIdx.x * PTS;

  _Float16* act  = (_Float16*)(smem_raw + ACT_OFF);   // [NCOL][KSTR]
  float*    qbuf = (float*)(smem_raw + QBUF_OFF);     // [NCOL][QSTR]
  float*    s2v  = (float*)(smem_raw + S2V_OFF);      // [128]
  float*    w3   = (float*)(smem_raw + W3_OFF);       // [2][128]

  const float scl  = scale[b];
  const float rscl = 1.0f / scl;
  const float zs0  = z_scalar[b * 2 + 0];
  const float zs1  = z_scalar[b * 2 + 1];

  // ---------------- Phase 0: stage input tile ----------------
  // point_feat rows k = 0..127 ; column n = 3*p + d ; float4 along m
  for (int idx = tid; idx < Cn * 3 * (PTS / 4); idx += 256) {
    const int g = idx & (PTS / 4 - 1);   // group of 4 points
    const int r = idx >> 3;
    const int d = r % 3;
    const int c = r / 3;
    const float4 v = *(const float4*)(
        point_feat + (((size_t)b * Cn + c) * 3 + d) * Mn + m0 + g * 4);
    const int p0 = g * 4;
    act[(3 * (p0 + 0) + d) * KSTR + c] = (_Float16)v.x;
    act[(3 * (p0 + 1) + d) * KSTR + c] = (_Float16)v.y;
    act[(3 * (p0 + 2) + d) * KSTR + c] = (_Float16)v.z;
    act[(3 * (p0 + 3) + d) * KSTR + c] = (_Float16)v.w;
  }
  // z_state rows k = 128..135 + zero pad k = 136..167
  for (int idx = tid; idx < NCOL * 40; idx += 256) {
    const int n  = idx / 40;
    const int kk = 128 + idx - n * 40;
    float v = 0.0f;
    if (kk < VIN) {
      const int j = kk - 128;
      const int d = n % 3;
      if (j < 2) v = (z_pos[(b * 2 + j) * 3 + d] - center[b * 3 + d]) * rscl;
      else       v = z_dir[(b * 6 + (j - 2)) * 3 + d];
    }
    act[n * KSTR + kk] = (_Float16)v;
  }
  if (tid < 128) s2v[tid] = W1x_s[tid * 2 + 0] * zs0 + W1x_s[tid * 2 + 1] * zs1;
  if (tid < 256) w3[tid] = W3x[tid];
  __syncthreads();

  // ---------------- Layer 1: q = W1x @ z ----------------
  gemm_tile<K1PAD / 32>(ws + W1H_OFF, K1PAD, act, qbuf, tid);
  __syncthreads();
  // q = q + s2v * q_hat  == q * (1 + s2v/(||q||+eps)) per (o, point) triple
  for (int idx = tid; idx < Hn * PTS; idx += 256) {
    const int o = idx & (Hn - 1);
    const int p = idx >> 7;
    const float q0 = qbuf[(3 * p + 0) * QSTR + o];
    const float q1 = qbuf[(3 * p + 1) * QSTR + o];
    const float q2 = qbuf[(3 * p + 2) * QSTR + o];
    const float nrm = sqrtf(q0 * q0 + q1 * q1 + q2 * q2) + EPSF;
    const float f = 1.0f + s2v[o] / nrm;
    act[(3 * p + 0) * KSTR + o] = (_Float16)(q0 * f);
    act[(3 * p + 1) * KSTR + o] = (_Float16)(q1 * f);
    act[(3 * p + 2) * KSTR + o] = (_Float16)(q2 * f);
  }
  __syncthreads();

  // d = W1x_d @ q ; h = q - min(q.d_hat, 0) * d_hat
  gemm_tile<Hn / 32>(ws + W1DH_OFF, Hn, act, qbuf, tid);
  __syncthreads();
  for (int idx = tid; idx < Hn * PTS; idx += 256) {
    const int o = idx & (Hn - 1);
    const int p = idx >> 7;
    const float d0 = qbuf[(3 * p + 0) * QSTR + o];
    const float d1 = qbuf[(3 * p + 1) * QSTR + o];
    const float d2 = qbuf[(3 * p + 2) * QSTR + o];
    const float q0 = (float)act[(3 * p + 0) * KSTR + o];
    const float q1 = (float)act[(3 * p + 1) * KSTR + o];
    const float q2 = (float)act[(3 * p + 2) * KSTR + o];
    const float inv = 1.0f / (sqrtf(d0 * d0 + d1 * d1 + d2 * d2) + EPSF);
    const float h0 = d0 * inv, h1 = d1 * inv, h2 = d2 * inv;
    const float c = fminf(q0 * h0 + q1 * h1 + q2 * h2, 0.0f);
    act[(3 * p + 0) * KSTR + o] = (_Float16)(q0 - c * h0);
    act[(3 * p + 1) * KSTR + o] = (_Float16)(q1 - c * h1);
    act[(3 * p + 2) * KSTR + o] = (_Float16)(q2 - c * h2);
  }
  __syncthreads();

  // ---------------- Layer 2: q = W2x @ h (no scalar inject) ----------------
  gemm_tile<Hn / 32>(ws + W2H_OFF, Hn, act, qbuf, tid);
  __syncthreads();
  for (int idx = tid; idx < Hn * PTS; idx += 256) {
    const int o = idx & (Hn - 1);
    const int p = idx >> 7;
    act[(3 * p + 0) * KSTR + o] = (_Float16)qbuf[(3 * p + 0) * QSTR + o];
    act[(3 * p + 1) * KSTR + o] = (_Float16)qbuf[(3 * p + 1) * QSTR + o];
    act[(3 * p + 2) * KSTR + o] = (_Float16)qbuf[(3 * p + 2) * QSTR + o];
  }
  __syncthreads();

  gemm_tile<Hn / 32>(ws + W2DH_OFF, Hn, act, qbuf, tid);
  __syncthreads();
  for (int idx = tid; idx < Hn * PTS; idx += 256) {
    const int o = idx & (Hn - 1);
    const int p = idx >> 7;
    const float d0 = qbuf[(3 * p + 0) * QSTR + o];
    const float d1 = qbuf[(3 * p + 1) * QSTR + o];
    const float d2 = qbuf[(3 * p + 2) * QSTR + o];
    const float q0 = (float)act[(3 * p + 0) * KSTR + o];
    const float q1 = (float)act[(3 * p + 1) * KSTR + o];
    const float q2 = (float)act[(3 * p + 2) * KSTR + o];
    const float inv = 1.0f / (sqrtf(d0 * d0 + d1 * d1 + d2 * d2) + EPSF);
    const float h0 = d0 * inv, h1 = d1 * inv, h2 = d2 * inv;
    const float c = fminf(q0 * h0 + q1 * h1 + q2 * h2, 0.0f);
    act[(3 * p + 0) * KSTR + o] = (_Float16)(q0 - c * h0);
    act[(3 * p + 1) * KSTR + o] = (_Float16)(q1 - c * h1);
    act[(3 * p + 2) * KSTR + o] = (_Float16)(q2 - c * h2);
  }
  __syncthreads();

  // ---------------- Head: v = W3x @ h2 ; write offsets (B,M,6) ----------------
  for (int idx = tid; idx < 6 * PTS; idx += 256) {
    const int p = idx / 6;
    const int j = idx - p * 6;
    const int o = j / 3;
    const int d = j - o * 3;
    const _Float16* arow = act + (3 * p + d) * KSTR;
    const float* wr = w3 + o * 128;
    float s = 0.0f;
#pragma unroll 8
    for (int k = 0; k < Hn; ++k) s += wr[k] * (float)arow[k];
    if (j < 3) s = s * scl + center[b * 3 + d];
    out[((size_t)b * Mn + m0 + p) * 6 + j] = s;
  }
}

// =====================================================================
// Kernel 2: global branch + gripper + argmin gather + ac assembly
// one workgroup per batch element
// =====================================================================
__global__ __launch_bounds__(256) void sim3_aux_kernel(
    const float* __restrict__ pc,        // (B,M,3)
    const float* __restrict__ z_pos, const float* __restrict__ z_dir,
    const float* __restrict__ z_scalar,
    const float* __restrict__ global_feat,  // (B,128,3)
    const float* __restrict__ center, const float* __restrict__ scale,
    const float* __restrict__ W1g,   const float* __restrict__ W1g_d,
    const float* __restrict__ W1g_s, const float* __restrict__ W2g,
    const float* __restrict__ W2g_d, const float* __restrict__ Wg_vs,
    float* __restrict__ out) {
  __shared__ float vg[VIN * 3];
  __shared__ float qA[Hn * 3];
  __shared__ float qB[Hn * 3];
  __shared__ float s2vg[Hn];
  __shared__ float zpn[6];
  __shared__ float grip[2];
  __shared__ float rdist[256];
  __shared__ int   ridx[256];

  const int tid = threadIdx.x;
  const int b   = blockIdx.x;
  const float rscl = 1.0f / scale[b];
  const float zs0 = z_scalar[b * 2 + 0];
  const float zs1 = z_scalar[b * 2 + 1];

  for (int idx = tid; idx < Cn * 3; idx += 256)
    vg[idx] = global_feat[(size_t)b * Cn * 3 + idx];
  if (tid < 24) {
    const int j = tid / 3, d = tid - j * 3;
    float v;
    if (j < 2) {
      v = (z_pos[(b * 2 + j) * 3 + d] - center[b * 3 + d]) * rscl;
      zpn[j * 3 + d] = v;
    } else {
      v = z_dir[(b * 6 + (j - 2)) * 3 + d];
    }
    vg[(128 + j) * 3 + d] = v;
  }
  if (tid < Hn) s2vg[tid] = W1g_s[tid * 2 + 0] * zs0 + W1g_s[tid * 2 + 1] * zs1;
  __syncthreads();

  // q = W1g @ vg
  for (int idx = tid; idx < Hn * 3; idx += 256) {
    const int o = idx / 3, d = idx - o * 3;
    float s = 0.0f;
    for (int k = 0; k < VIN; ++k) s += W1g[o * VIN + k] * vg[k * 3 + d];
    qA[o * 3 + d] = s;
  }
  __syncthreads();
  if (tid < Hn) {
    const float q0 = qA[tid * 3], q1 = qA[tid * 3 + 1], q2 = qA[tid * 3 + 2];
    const float f = 1.0f + s2vg[tid] / (sqrtf(q0 * q0 + q1 * q1 + q2 * q2) + EPSF);
    qA[tid * 3] = q0 * f; qA[tid * 3 + 1] = q1 * f; qA[tid * 3 + 2] = q2 * f;
  }
  __syncthreads();
  for (int idx = tid; idx < Hn * 3; idx += 256) {
    const int o = idx / 3, d = idx - o * 3;
    float s = 0.0f;
    for (int k = 0; k < Hn; ++k) s += W1g_d[o * Hn + k] * qA[k * 3 + d];
    qB[o * 3 + d] = s;
  }
  __syncthreads();
  if (tid < Hn) {
    const float d0 = qB[tid * 3], d1 = qB[tid * 3 + 1], d2 = qB[tid * 3 + 2];
    const float q0 = qA[tid * 3], q1 = qA[tid * 3 + 1], q2 = qA[tid * 3 + 2];
    const float inv = 1.0f / (sqrtf(d0 * d0 + d1 * d1 + d2 * d2) + EPSF);
    const float h0 = d0 * inv, h1 = d1 * inv, h2 = d2 * inv;
    const float c = fminf(q0 * h0 + q1 * h1 + q2 * h2, 0.0f);
    qA[tid * 3] = q0 - c * h0; qA[tid * 3 + 1] = q1 - c * h1; qA[tid * 3 + 2] = q2 - c * h2;
  }
  __syncthreads();
  // layer 2
  for (int idx = tid; idx < Hn * 3; idx += 256) {
    const int o = idx / 3, d = idx - o * 3;
    float s = 0.0f;
    for (int k = 0; k < Hn; ++k) s += W2g[o * Hn + k] * qA[k * 3 + d];
    qB[o * 3 + d] = s;
  }
  __syncthreads();
  for (int idx = tid; idx < Hn * 3; idx += 256) {
    const int o = idx / 3, d = idx - o * 3;
    float s = 0.0f;
    for (int k = 0; k < Hn; ++k) s += W2g_d[o * Hn + k] * qB[k * 3 + d];
    vg[o * 3 + d] = s;   // reuse vg as scratch
  }
  __syncthreads();
  if (tid < Hn) {
    const float d0 = vg[tid * 3], d1 = vg[tid * 3 + 1], d2 = vg[tid * 3 + 2];
    const float q0 = qB[tid * 3], q1 = qB[tid * 3 + 1], q2 = qB[tid * 3 + 2];
    const float inv = 1.0f / (sqrtf(d0 * d0 + d1 * d1 + d2 * d2) + EPSF);
    const float h0 = d0 * inv, h1 = d1 * inv, h2 = d2 * inv;
    const float c = fminf(q0 * h0 + q1 * h1 + q2 * h2, 0.0f);
    qB[tid * 3] = q0 - c * h0; qB[tid * 3 + 1] = q1 - c * h1; qB[tid * 3 + 2] = q2 - c * h2;
  }
  __syncthreads();
  if (tid < 2) {
    float g = 0.0f;
    for (int k = 0; k < Hn; ++k) {
      const float a0 = qB[k * 3], a1 = qB[k * 3 + 1], a2 = qB[k * 3 + 2];
      g += Wg_vs[tid * Hn + k] * sqrtf(a0 * a0 + a1 * a1 + a2 * a2);
    }
    const float ga = 1.0f / (1.0f + expf(-g));
    grip[tid] = ga;
    out[OFF_GRIP + b * 2 + tid] = ga;
  }
  __syncthreads();

  // argmin over pc per embedding e, then gather offsets row -> ac
  for (int e = 0; e < 2; ++e) {
    const float x = zpn[e * 3], y = zpn[e * 3 + 1], z = zpn[e * 3 + 2];
    float best = 3.4e38f; int bi = 0;
    for (int m = tid; m < Mn; m += 256) {
      const float dx = x - pc[((size_t)b * Mn + m) * 3 + 0];
      const float dy = y - pc[((size_t)b * Mn + m) * 3 + 1];
      const float dz = z - pc[((size_t)b * Mn + m) * 3 + 2];
      const float d2 = dx * dx + dy * dy + dz * dz;
      if (d2 < best) { best = d2; bi = m; }
    }
    rdist[tid] = best; ridx[tid] = bi;
    __syncthreads();
    if (tid == 0) {
      float bb = rdist[0]; int bj = ridx[0];
      for (int t = 1; t < 256; ++t) {
        if (rdist[t] < bb || (rdist[t] == bb && ridx[t] < bj)) { bb = rdist[t]; bj = ridx[t]; }
      }
      const size_t base = ((size_t)b * Mn + bj) * 6;
      float* ac = out + OFF_AC + b * 14 + e * 7;
      ac[0] = grip[e];
      for (int j = 0; j < 6; ++j) ac[1 + j] = out[base + j];
    }
    __syncthreads();
  }
}

// =====================================================================
extern "C" void kernel_launch(void* const* d_in, const int* in_sizes, int n_in,
                              void* d_out, int out_size, void* d_ws, size_t ws_size,
                              hipStream_t stream) {
  const float* pc          = (const float*)d_in[0];
  const float* z_pos       = (const float*)d_in[1];
  const float* z_dir       = (const float*)d_in[2];
  const float* z_scalar    = (const float*)d_in[3];
  const float* point_feat  = (const float*)d_in[4];
  const float* global_feat = (const float*)d_in[5];
  const float* center      = (const float*)d_in[6];
  const float* scale       = (const float*)d_in[7];
  const float* W1x         = (const float*)d_in[8];
  const float* W1x_d       = (const float*)d_in[9];
  const float* W1x_s       = (const float*)d_in[10];
  const float* W2x         = (const float*)d_in[11];
  const float* W2x_d       = (const float*)d_in[12];
  const float* W3x         = (const float*)d_in[13];
  const float* W1g         = (const float*)d_in[14];
  const float* W1g_d       = (const float*)d_in[15];
  const float* W1g_s       = (const float*)d_in[16];
  const float* W2g         = (const float*)d_in[17];
  const float* W2g_d       = (const float*)d_in[18];
  const float* Wg_vs       = (const float*)d_in[19];
  float* out = (float*)d_out;
  _Float16* wsh = (_Float16*)d_ws;

  // 0) weight precision prep into workspace (re-done every call; deterministic)
  sim3_prep_weights<<<(WS_HALVES + 255) / 256, 256, 0, stream>>>(
      W1x, W1x_d, W2x, W2x_d, wsh);

  // 1) fused point MLP: grid = (M/PTS, B)
  dim3 grid1(Mn / PTS, Bn);
  sim3_points_kernel<<<grid1, 256, LDS_TOTAL, stream>>>(
      point_feat, z_pos, z_dir, z_scalar, center, scale, W1x_s, W3x, wsh, out);

  // 2) global branch + argmin/gather + ac
  sim3_aux_kernel<<<Bn, 256, 0, stream>>>(
      pc, z_pos, z_dir, z_scalar, global_feat, center, scale,
      W1g, W1g_d, W1g_s, W2g, W2g_d, Wg_vs, out);

  (void)in_sizes; (void)n_in; (void)out_size; (void)ws_size;
}
